// QuantizingEncodecEncoder_50826642981375
// MI455X (gfx1250) — matrix operand order repaired and checked by reading, not compiled, not run on hardware
//
#include <hip/hip_runtime.h>
#include <hip/hip_bf16.h>

typedef __attribute__((ext_vector_type(16))) _Float16 v16h;
typedef __attribute__((ext_vector_type(8)))  _Float16 v8h;
typedef __attribute__((ext_vector_type(4)))  _Float16 v4h;
typedef __attribute__((ext_vector_type(8)))  float    v8f;

__device__ __forceinline__ float eluf(float x) { return x > 0.f ? x : (__expf(x) - 1.f); }
__device__ __forceinline__ float sigf(float x) { return 1.f / (1.f + __expf(-x)); }
__device__ __forceinline__ int refl(int i, int L) { if (i < 0) i = -i; if (i >= L) i = 2 * L - 2 - i; return i; }

// ---------------------------------------------------------------------------
// Generic conv-as-GEMM with WMMA f16 (f32 accumulate).
// Out[b][m][t] = sum_tap sum_k W[m][k][tap] * f(X[b][k][reflect(t*stride - pl + tap)]) + bias[m]
// Matrix view: M=Cout (A = weights), N = B*Tout (gathered input cols), K = Cin per tap.
// Block: 256 thr = 8 waves (2x4), tile 32(M) x 64(N), K in 32-chunks.
// LDS tiles live in WMMA *fragment order* [tile][lane][16]: each wave reads its
// whole fragment as one aligned v16h (2x ds_load_b128).
// GUARD=false instantiation (Cin%32==0, Cout%32==0, N%64==0) has fully
// unconditional staging loads; GUARD=true uses clamped addresses + post-load
// select (no per-element exec-mask branches either way).
// ---------------------------------------------------------------------------
template <bool GUARD, bool ELU>
__global__ __launch_bounds__(256) void conv_gemm_kernel(
    const float* __restrict__ X, const float* __restrict__ W, const float* __restrict__ bias,
    float* __restrict__ Out, int Bsz, int Cin, int Tin, int Cout, int Tout,
    int ktaps, int stride, int pl, int accum)
{
  __shared__ __align__(32) _Float16 Asp[2 * 32 * 16]; // A fragments: [mt][lane][i]
  __shared__ __align__(32) _Float16 Bsp[4 * 32 * 16]; // B fragments: [nt][lane][i]
  const int tid  = threadIdx.x;
  const int lane = tid & 31;
  const int wave = tid >> 5;
  const int mBlk = blockIdx.y * 32;
  const int nBlk = blockIdx.x * 64;
  const int N = Bsz * Tout;
  const int nkc = (Cin + 31) >> 5;

  // staging coordinates (fixed per thread)
  const int abase   = tid * 4;                 // A: 1024 halfs / 256 thr
  const int a_i0    = abase & 15;              // 0,4,8,12
  const int a_lane  = (abase >> 4) & 31;
  const int a_mt    = abase >> 9;              // 0..1
  const int a_m     = mBlk + a_mt * 16 + (a_lane & 15);
  const int a_mcl   = GUARD ? (a_m < Cout ? a_m : Cout - 1) : a_m;
  const int a_khalf = a_lane >> 4;

  const int bbase   = tid * 8;                 // B: 2048 halfs / 256 thr
  const int b_i0    = bbase & 15;              // 0 or 8
  const int b_lane  = (bbase >> 4) & 31;
  const int b_nt    = bbase >> 9;              // 0..3
  const int b_n     = nBlk + b_nt * 16 + (b_lane & 15);
  const int b_khalf = b_lane >> 4;
  const bool b_valid = GUARD ? (b_n < N) : true;
  const int b_ncl   = GUARD ? (b_valid ? b_n : N - 1) : b_n;
  const int b_b     = b_ncl / Tout;
  const int b_t     = b_ncl % Tout;

  v8f acc = {};
  for (int tap = 0; tap < ktaps; ++tap) {
    const int b_src = refl(b_t * stride - pl + tap, Tin); // same for all k of this thread
    for (int kc = 0; kc < nkc; ++kc) {
      __syncthreads();
      // ---- stage A (weights) in fragment order: 4 unconditional loads -> 1 b64 store
      {
        float vv[4];
        #pragma unroll
        for (int u = 0; u < 4; ++u) {
          int i = a_i0 + u;
          int k = (kc << 5) + (a_khalf << 4) + i;                        // A K = 16*half + i
          int kcl = GUARD ? (k < Cin ? k : Cin - 1) : k;
          float v = W[(a_mcl * Cin + kcl) * ktaps + tap];
          if (GUARD) v = (a_m < Cout && k < Cin) ? v : 0.f;
          vv[u] = v;
        }
        v4h hv;
        #pragma unroll
        for (int u = 0; u < 4; ++u) hv[u] = (_Float16)vv[u];
        *(v4h*)&Asp[abase] = hv;
      }
      // ---- stage B (input cols) in fragment order: 8 unconditional loads -> 1 b128 store
      {
        float vv[8];
        #pragma unroll
        for (int u = 0; u < 8; ++u) {
          int k = (kc << 5) + (b_khalf << 4) + b_i0 + u;                 // B K = 16*half + i
          int kcl = GUARD ? (k < Cin ? k : Cin - 1) : k;
          float v = X[(b_b * Cin + kcl) * Tin + b_src];
          if (ELU) v = eluf(v);
          if (GUARD) v = (b_valid && k < Cin) ? v : 0.f;
          vv[u] = v;
        }
        v8h hv;
        #pragma unroll
        for (int u = 0; u < 8; ++u) hv[u] = (_Float16)vv[u];
        *(v8h*)&Bsp[bbase] = hv;
      }
      __syncthreads();
      // ---- whole-fragment LDS reads (2x ds_load_b128 each) + WMMA
      v16h af = *(const v16h*)&Asp[((wave >> 2) * 32 + lane) * 16];
      v16h bf = *(const v16h*)&Bsp[((wave & 3) * 32 + lane) * 16];
      acc = __builtin_amdgcn_wmma_f32_16x16x32_f16(false, af, false, bf, (short)0, acc,
                                                   false, false);
    }
  }
  // store D (16x16 f32 layout: lane<16 -> M=v, lane>=16 -> M=v+8)
  const int n_g = nBlk + (wave & 3) * 16 + (lane & 15);
  const int mh  = (lane >> 4) * 8;
  #pragma unroll
  for (int v = 0; v < 8; ++v) {
    int m_g = mBlk + (wave >> 2) * 16 + mh + v;
    if (m_g < Cout && n_g < N) {
      int b = n_g / Tout, t = n_g % Tout;
      float val = acc[v] + bias[m_g];
      int o = (b * Cout + m_g) * Tout + t;
      if (accum) Out[o] += val; else Out[o] = val;
    }
  }
}

// ---------------------------------------------------------------------------
// Pack W (N x K, row-major, B^T view) into WMMA B-fragment order:
// P[(((tile*KC + kchunk)*32 + lane)*16 + i)] = (f16) W[n][k],
//   n = tile*16 + (lane&15), k = kchunk*32 + (lane>>4)*16 + i.
// ---------------------------------------------------------------------------
__global__ void pack_b_kernel(const float* __restrict__ W, _Float16* __restrict__ P, int N, int K)
{
  int o = blockIdx.x * blockDim.x + threadIdx.x;
  int KC = K >> 5;
  int total = (N >> 4) * KC * 512;
  if (o >= total) return;
  int i = o & 15;
  int lane = (o >> 4) & 31;
  int rest = o >> 9;
  int kchunk = rest % KC;
  int tile = rest / KC;
  int n = (tile << 4) + (lane & 15);
  int k = (kchunk << 5) + ((lane >> 4) << 4) + i;
  P[o] = (_Float16)W[n * K + k];
}

// ---------------------------------------------------------------------------
// LSTM scan: one workgroup (32 waves). Wave w owns hidden slice [16w, 16w+16)
// and its 4 gate tiles (i,f,g,o at columns g*512 + 16w). h kept in LDS packed
// as A-fragments (16 rows: batch 0..3 valid, rest provably stay zero).
// Whh^T pre-packed (pack_b_kernel) -> contiguous v16h global loads.
// c kept in VGPRs. 64 v_wmma per wave per timestep.
// ---------------------------------------------------------------------------
__global__ __launch_bounds__(1024) void lstm_scan_kernel(
    const float* __restrict__ xp,      // (4, 2048, T)
    const _Float16* __restrict__ whhp, // packed (128 tiles, 16 kchunks, 32, 16)
    float* __restrict__ y,             // (4, 512, T)
    int T)
{
  __shared__ __align__(32) _Float16 hbuf[16 * 32 * 16]; // A-fragments of h (16x512)
  const int tid  = threadIdx.x;
  const int lane = tid & 31;
  const int wave = tid >> 5;
  const int h0 = wave * 16;
  for (int e = tid; e < 16 * 32 * 16; e += 1024) hbuf[e] = (_Float16)0.f;
  v8f c = {};
  __syncthreads();

  const int nl = lane & 15;
  const int mh = (lane >> 4) * 8;
  const int j = h0 + nl;                        // hidden column owned by this lane
  const int kchunk_j = j >> 5;                  // h-write coords (A-fragment inverse map)
  const int kcm = j & 31;
  const int khalf_j = kcm >> 4;
  const int i_j = kcm & 15;

  for (int t = 0; t < T; ++t) {
    v8f a0, a1, a2, a3; // accumulators init = x_proj fragments (gate i,f,g,o)
    #pragma unroll
    for (int v = 0; v < 8; ++v) {
      int m = mh + v;
      int mcl = m < 4 ? m : 3;                  // clamp + select: unconditional loads
      float x0 = xp[(mcl * 2048 + (0 * 512 + j)) * T + t];
      float x1 = xp[(mcl * 2048 + (1 * 512 + j)) * T + t];
      float x2 = xp[(mcl * 2048 + (2 * 512 + j)) * T + t];
      float x3 = xp[(mcl * 2048 + (3 * 512 + j)) * T + t];
      a0[v] = (m < 4) ? x0 : 0.f;
      a1[v] = (m < 4) ? x1 : 0.f;
      a2[v] = (m < 4) ? x2 : 0.f;
      a3[v] = (m < 4) ? x3 : 0.f;
    }
    #pragma unroll 4
    for (int kc = 0; kc < 16; ++kc) {
      v16h af = *(const v16h*)&hbuf[(kc * 32 + lane) * 16];
      v16h b0 = *(const v16h*)&whhp[(((0 * 32 + wave) * 16 + kc) * 32 + lane) * 16];
      v16h b1 = *(const v16h*)&whhp[(((1 * 32 + wave) * 16 + kc) * 32 + lane) * 16];
      v16h b2 = *(const v16h*)&whhp[(((2 * 32 + wave) * 16 + kc) * 32 + lane) * 16];
      v16h b3 = *(const v16h*)&whhp[(((3 * 32 + wave) * 16 + kc) * 32 + lane) * 16];
      a0 = __builtin_amdgcn_wmma_f32_16x16x32_f16(false, af, false, b0, (short)0, a0, false, false);
      a1 = __builtin_amdgcn_wmma_f32_16x16x32_f16(false, af, false, b1, (short)0, a1, false, false);
      a2 = __builtin_amdgcn_wmma_f32_16x16x32_f16(false, af, false, b2, (short)0, a2, false, false);
      a3 = __builtin_amdgcn_wmma_f32_16x16x32_f16(false, af, false, b3, (short)0, a3, false, false);
    }
    float hv[8];
    #pragma unroll
    for (int v = 0; v < 8; ++v) {
      float iv = sigf(a0[v]);
      float fv = sigf(a1[v]);
      float gv = tanhf(a2[v]);
      float ov = sigf(a3[v]);
      float cv = fv * c[v] + iv * gv;
      c[v] = cv;
      hv[v] = ov * tanhf(cv);
    }
    __syncthreads(); // all waves done reading hbuf for this step
    #pragma unroll
    for (int v = 0; v < 8; ++v) {
      int m = mh + v;
      hbuf[(kchunk_j * 32 + (m + (khalf_j << 4))) * 16 + i_j] = (_Float16)hv[v];
    }
    if (lane < 16) {
      #pragma unroll
      for (int v = 0; v < 4; ++v)
        y[(v * 512 + j) * T + t] = hv[v]; // lane<16 => batch row m=v
    }
    __syncthreads();
  }
}

// ---------------------------------------------------------------------------
// Small helper kernels
// ---------------------------------------------------------------------------
__global__ void conv0_kernel(const float* __restrict__ x, const float* __restrict__ w,
                             const float* __restrict__ bias, float* __restrict__ out, int T)
{
  int e = blockIdx.x * blockDim.x + threadIdx.x;
  if (e >= 4 * 32 * T) return;
  int t = e % T; int o = (e / T) % 32; int b = e / (32 * T);
  float acc = bias[o];
  #pragma unroll
  for (int k = 0; k < 7; ++k) acc += x[b * T + refl(t - 3 + k, T)] * w[o * 7 + k];
  out[e] = acc;
}

__global__ void bias_comb_kernel(const float* a, const float* b, float* o, int n)
{ int i = blockIdx.x * blockDim.x + threadIdx.x; if (i < n) o[i] = a[i] + b[i]; }

__global__ void add_kernel(float* a, const float* b, int n)
{ int i = blockIdx.x * blockDim.x + threadIdx.x; if (i < n) a[i] += b[i]; }

__global__ void zero_kernel(float* p, int n)
{ int i = blockIdx.x * blockDim.x + threadIdx.x; if (i < n) p[i] = 0.f; }

__global__ void cbnorm_kernel(const float* __restrict__ cb, float* __restrict__ bias, int bins, int D)
{
  int j = blockIdx.x * blockDim.x + threadIdx.x;
  if (j >= bins) return;
  float s = 0.f;
  for (int d = 0; d < D; ++d) { float v = cb[j * D + d]; s += v * v; }
  bias[j] = -0.5f * s;
}

// argmax of scores == argmin of L2 distance (tie -> lowest index, matching argmin)
__global__ __launch_bounds__(256) void argmax_kernel(const float* __restrict__ S,
                                                     int* __restrict__ idx, int T)
{
  __shared__ float bv[256];
  __shared__ int   bi[256];
  int n = blockIdx.x; int b = n / T, t = n % T;
  float best = -3.4e38f; int besti = 0;
  for (int j = threadIdx.x; j < 1024; j += 256) {
    float v = S[(b * 1024 + j) * T + t];
    if (v > best) { best = v; besti = j; }
  }
  bv[threadIdx.x] = best; bi[threadIdx.x] = besti;
  __syncthreads();
  for (int s2 = 128; s2 > 0; s2 >>= 1) {
    if (threadIdx.x < s2) {
      float v2 = bv[threadIdx.x + s2]; int i2 = bi[threadIdx.x + s2];
      if (v2 > bv[threadIdx.x] || (v2 == bv[threadIdx.x] && i2 < bi[threadIdx.x])) {
        bv[threadIdx.x] = v2; bi[threadIdx.x] = i2;
      }
    }
    __syncthreads();
  }
  if (threadIdx.x == 0) idx[n] = bi[0];
}

__global__ void rvq_update_kernel(const float* __restrict__ cb, const int* __restrict__ idx,
                                  float* __restrict__ resid, float* __restrict__ qout, int T)
{
  int e = blockIdx.x * blockDim.x + threadIdx.x;
  if (e >= 4 * T * 128) return;
  int d = e & 127; int n = e >> 7; int b = n / T, t = n % T;
  float v = cb[idx[n] * 128 + d];
  int o = (b * 128 + d) * T + t;
  resid[o] -= v;
  qout[o]  += v;
}

// ---------------------------------------------------------------------------
// Host-side orchestration
// ---------------------------------------------------------------------------
static void launch_conv_gemm(const float* X, const float* W, const float* bias, float* Out,
                             int Bsz, int Cin, int Tin, int Cout, int Tout,
                             int ktaps, int stride, int pl, int elu_in, int accum,
                             hipStream_t stream)
{
  int N = Bsz * Tout;
  dim3 grid((N + 63) / 64, (Cout + 31) / 32);
  bool guard = (Cin & 31) || (Cout & 31) || (N & 63);
  if (guard) {
    if (elu_in)
      conv_gemm_kernel<true, true><<<grid, 256, 0, stream>>>(X, W, bias, Out, Bsz, Cin, Tin,
                                                             Cout, Tout, ktaps, stride, pl, accum);
    else
      conv_gemm_kernel<true, false><<<grid, 256, 0, stream>>>(X, W, bias, Out, Bsz, Cin, Tin,
                                                              Cout, Tout, ktaps, stride, pl, accum);
  } else {
    if (elu_in)
      conv_gemm_kernel<false, true><<<grid, 256, 0, stream>>>(X, W, bias, Out, Bsz, Cin, Tin,
                                                              Cout, Tout, ktaps, stride, pl, accum);
    else
      conv_gemm_kernel<false, false><<<grid, 256, 0, stream>>>(X, W, bias, Out, Bsz, Cin, Tin,
                                                               Cout, Tout, ktaps, stride, pl, accum);
  }
}

extern "C" void kernel_launch(void* const* d_in, const int* in_sizes, int n_in,
                              void* d_out, int out_size, void* d_ws, size_t ws_size,
                              hipStream_t stream)
{
  (void)in_sizes; (void)n_in; (void)out_size; (void)ws_size;
  // ---- inputs (setup_inputs dict order) ----
  const float* xin = (const float*)d_in[0];
  const float* c0w = (const float*)d_in[1];
  const float* c0b = (const float*)d_in[2];
  const float *c1w[4], *c1b[4], *c2w[4], *c2b[4], *scw[4], *scb[4], *dww[4], *dwb[4];
  int p = 3;
  for (int s = 0; s < 4; ++s) {
    c1w[s] = (const float*)d_in[p++]; c1b[s] = (const float*)d_in[p++];
    c2w[s] = (const float*)d_in[p++]; c2b[s] = (const float*)d_in[p++];
    scw[s] = (const float*)d_in[p++]; scb[s] = (const float*)d_in[p++];
    dww[s] = (const float*)d_in[p++]; dwb[s] = (const float*)d_in[p++];
  }
  const float *Wih[2], *Whh[2], *bih[2], *bhh[2];
  for (int l = 0; l < 2; ++l) {
    Wih[l] = (const float*)d_in[p++]; Whh[l] = (const float*)d_in[p++];
    bih[l] = (const float*)d_in[p++]; bhh[l] = (const float*)d_in[p++];
  }
  const float* cfw = (const float*)d_in[p++];
  const float* cfb = (const float*)d_in[p++];
  const float* cbs = (const float*)d_in[p++];
  float* dout = (float*)d_out;

  // ---- workspace layout (floats) ----
  float* ws   = (float*)d_ws;
  float* bufA = ws;                  // 30,720,000
  float* bufB = ws + 30720000;       // 30,720,000
  float* bufC = ws + 61440000;       // 15,360,000
  float* Sml  = ws + 76800000;       // small region
  float* Y1   = Sml;                 // 1,536,000
  float* Y2   = Sml + 1600000;       // 1,536,000
  _Float16* PW = (_Float16*)(Sml + 3200000); // 1,048,576 halfs (32B-aligned offset)
  float* CBB  = Sml + 3800000;       // 1024
  int*   IDX  = (int*)(Sml + 3810000); // 3000
  float* BIASC = Sml + 3820000;      // 2048

  const int Cs[4] = {32, 64, 128, 256};
  const int rs[4] = {2, 4, 5, 8};

  // ---- conv0: 1 -> 32, k=7, reflect pad 3/3 ----
  {
    int total = 4 * 32 * 240000;
    conv0_kernel<<<(total + 255) / 256, 256, 0, stream>>>(xin, c0w, c0b, bufA, 240000);
  }

  // ---- SEANet stages ----
  int T = 240000;
  for (int s = 0; s < 4; ++s) {
    int C = Cs[s], H = C / 2, r = rs[s];
    // u1 = conv3(elu(h))           : bufA -> bufC
    launch_conv_gemm(bufA, c1w[s], c1b[s], bufC, 4, C, T, H, T, 3, 1, 1, 1, 0, stream);
    // u2 = conv1(elu(u1))          : bufC -> bufB
    launch_conv_gemm(bufC, c2w[s], c2b[s], bufB, 4, H, T, C, T, 1, 1, 0, 1, 0, stream);
    // h  = shortcut(h) + u2        : bufA -> bufB (accumulate)
    launch_conv_gemm(bufA, scw[s], scb[s], bufB, 4, C, T, C, T, 1, 1, 0, 0, 1, stream);
    // down = conv(elu(h), k=2r, s=r): bufB -> bufA
    launch_conv_gemm(bufB, dww[s], dwb[s], bufA, 4, C, T, 2 * C, T / r,
                     2 * r, r, r - r / 2, 1, 0, stream);
    T /= r;
  }
  // h_enc in bufA: (4, 512, 750)

  // ---- LSTM layer 0 ----
  bias_comb_kernel<<<8, 256, 0, stream>>>(bih[0], bhh[0], BIASC, 2048);
  launch_conv_gemm(bufA, Wih[0], BIASC, bufB, 4, 512, 750, 2048, 750, 1, 1, 0, 0, 0, stream);
  pack_b_kernel<<<(2048 * 512 + 255) / 256, 256, 0, stream>>>(Whh[0], PW, 2048, 512);
  lstm_scan_kernel<<<1, 1024, 0, stream>>>(bufB, PW, Y1, 750);

  // ---- LSTM layer 1 ----
  bias_comb_kernel<<<8, 256, 0, stream>>>(bih[1], bhh[1], BIASC, 2048);
  launch_conv_gemm(Y1, Wih[1], BIASC, bufB, 4, 512, 750, 2048, 750, 1, 1, 0, 0, 0, stream);
  pack_b_kernel<<<(2048 * 512 + 255) / 256, 256, 0, stream>>>(Whh[1], PW, 2048, 512);
  lstm_scan_kernel<<<1, 1024, 0, stream>>>(bufB, PW, Y2, 750);

  // ---- residual skip + final conv ----
  add_kernel<<<(4 * 512 * 750 + 255) / 256, 256, 0, stream>>>(bufA, Y2, 4 * 512 * 750);
  launch_conv_gemm(bufA, cfw, cfb, bufB, 4, 512, 750, 128, 750, 7, 1, 3, 1, 0, stream);
  // z (residual) in bufB: (4, 128, 750)

  // ---- RVQ: 8 stages ----
  zero_kernel<<<(384000 + 255) / 256, 256, 0, stream>>>(dout, 384000);
  for (int q = 0; q < 8; ++q) {
    const float* cb = cbs + (size_t)q * 1024 * 128;
    cbnorm_kernel<<<4, 256, 0, stream>>>(cb, CBB, 1024, 128);
    // scores[b][j][t] = r.c_j - 0.5||c_j||^2   (argmax == argmin distance)
    launch_conv_gemm(bufB, cb, CBB, bufC, 4, 128, 750, 1024, 750, 1, 1, 0, 0, 0, stream);
    argmax_kernel<<<3000, 256, 0, stream>>>(bufC, IDX, 750);
    rvq_update_kernel<<<(3000 * 128 + 255) / 256, 256, 0, stream>>>(cb, IDX, bufB, dout, 750);
  }
}
